// Attention_20968030339767
// MI455X (gfx1250) — matrix operand order repaired and checked by reading, not compiled, not run on hardware
//
#include <hip/hip_runtime.h>

typedef __attribute__((ext_vector_type(16))) _Float16 v16h;
typedef __attribute__((ext_vector_type(8)))  float    v8f;
typedef __attribute__((ext_vector_type(4)))  int      v4i;

#define DIMC   256
#define HEADS  8
#define DKV    64
#define NTOK   1024
#define NB     16
#define INKV   512
#define OC3    1536
#define L2E    1.4426950408889634f

// LDS slab strides (halves) — padded to spread banks
#define KLD    72
#define VLD    40

// Try the CDNA5 async global->LDS path; fall back to load+ds_store staging.
#ifndef USE_ASYNC_LDS
#if defined(__has_builtin)
#if __has_builtin(__builtin_amdgcn_global_load_async_to_lds_b128) && \
    __has_builtin(__builtin_amdgcn_s_wait_asynccnt)
#define USE_ASYNC_LDS 1
#endif
#endif
#endif
#ifndef USE_ASYNC_LDS
#define USE_ASYNC_LDS 0
#endif

// ---------------- workspace layout (bytes) ----------------
static constexpr size_t OFF_XH  = 0;                                        // f16 [B][N][DIM]
static constexpr size_t OFF_WF  = OFF_XH  + (size_t)NB*NTOK*DIMC*2;         // f16 [1536][256]
static constexpr size_t OFF_CF  = OFF_WF  + (size_t)OC3*DIMC*2;             // f32 [1536]
static constexpr size_t OFF_WOF = OFF_CF  + (size_t)OC3*4;                  // f16 [256][512]
static constexpr size_t OFF_COF = OFF_WOF + (size_t)DIMC*INKV*2;            // f32 [256]
static constexpr size_t OFF_Q   = OFF_COF + (size_t)DIMC*4;                 // f16 [B][H][N][64]
static constexpr size_t OFF_K   = OFF_Q   + (size_t)NB*HEADS*NTOK*DKV*2;
static constexpr size_t OFF_VT  = OFF_K   + (size_t)NB*HEADS*NTOK*DKV*2;    // f16 [B][H][64][N]
static constexpr size_t OFF_HS  = OFF_VT  + (size_t)NB*HEADS*DKV*NTOK*2;    // f16 [B][N][512]

// ---------------- helpers ----------------
union ABu { v16h v; uint4 q[2]; unsigned int u[8]; };

__device__ __forceinline__ float fexp2(float x) {
#if __has_builtin(__builtin_amdgcn_exp2f)
  return __builtin_amdgcn_exp2f(x);
#else
  return exp2f(x);
#endif
}

__device__ __forceinline__ unsigned packh2(float a, float b) {
  union { _Float16 h[2]; unsigned u; } x;
  x.h[0] = (_Float16)a; x.h[1] = (_Float16)b; return x.u;
}

// A-operand (16x32 f16): lane<16 -> row M=lane, halves = K kb..kb+15
//                        lane>=16 -> row M=lane-16, halves = K kb+16..kb+31
__device__ __forceinline__ v16h load_a(const _Float16* __restrict__ src, int ld,
                                       int mbase, int kb, int lane) {
  const int lm  = lane & 15;
  const int hi8 = (lane >> 4) << 4;
  const _Float16* p = src + (size_t)(mbase + lm) * ld + kb + hi8;
  ABu a;
  a.q[0] = *(const uint4*)(p);
  a.q[1] = *(const uint4*)(p + 8);
  return a.v;
}

// B-operand (32x16 f16): lane<16 -> col N=lane, halves = K kb..kb+15
//                        lane>=16 -> col N=lane-16, halves = K kb+16..kb+31
__device__ __forceinline__ v16h load_b(const _Float16* __restrict__ src, int ld,
                                       int nbase, int kb, int lane) {
  const int ln  = lane & 15;
  const int k16 = (lane >> 4) << 4;
  const _Float16* p = src + (size_t)(nbase + ln) * ld + kb + k16;
  ABu b;
  b.q[0] = *(const uint4*)(p);
  b.q[1] = *(const uint4*)(p + 8);
  return b.v;
}

__device__ __forceinline__ v8f wmma_f16(v16h a, v16h b, v8f c) {
  return __builtin_amdgcn_wmma_f32_16x16x32_f16(false, a, false, b, (short)0, c, false, false);
}

// Stage one 32-j slab of K [32 x 64] and V^T [64 x 32] into LDS.
// 256 threads: t<128 move K (16B each), t>=128 move V^T (16B each).
__device__ __forceinline__ void stage_slab(const _Float16* __restrict__ kg,
                                           const _Float16* __restrict__ vg,
                                           _Float16* lk, _Float16* lv,
                                           int jb, int t) {
  const _Float16* g;
  _Float16* l;
  if (t < 128) {
    const int row = t >> 2, ch = (t & 3) << 4;
    g = kg + (size_t)(jb + row) * DKV + ch;
    l = lk + row * KLD + ch;
  } else {
    const int u = t & 127;
    const int d = u >> 1, ch = (u & 1) << 4;
    g = vg + (size_t)d * NTOK + jb + ch;
    l = lv + d * VLD + ch;
  }
#if USE_ASYNC_LDS
  __builtin_amdgcn_global_load_async_to_lds_b128(
      (__attribute__((address_space(1))) v4i*)g,
      (__attribute__((address_space(3))) v4i*)l, 0, 0);
#else
  *(uint4*)l = *(const uint4*)g;
#endif
}

// ---------------- prep: x[b][c][n] f32 -> xh[b][n][c] f16 (LDS tiled transpose) ----
__global__ __launch_bounds__(256) void prep_xt(const float* __restrict__ x,
                                               _Float16* __restrict__ xh) {
  __shared__ float tile[32][33];
  const int b = blockIdx.z, cb = blockIdx.x * 32, nb = blockIdx.y * 32;
  const int tx = threadIdx.x, ty = threadIdx.y;        // (32, 8)
  const float* xb = x + ((size_t)b * DIMC + cb) * NTOK + nb;
#pragma unroll
  for (int k = 0; k < 4; ++k)
    tile[ty + 8 * k][tx] = xb[(size_t)(ty + 8 * k) * NTOK + tx];
  __syncthreads();
  _Float16* xo = xh + ((size_t)b * NTOK + nb) * DIMC + cb;
#pragma unroll
  for (int k = 0; k < 4; ++k)
    xo[(size_t)(ty + 8 * k) * DIMC + tx] = (_Float16)tile[tx][ty + 8 * k];
}

// ---------------- prep: fold BN (+dot-scale into Wq) into f16 weights ----------
__global__ __launch_bounds__(256) void prep_wqkv(
    const float* wq, const float* gq, const float* bq, const float* mq, const float* vq,
    const float* wk, const float* gk, const float* bk, const float* mk, const float* vk,
    const float* wv, const float* gv, const float* bv, const float* mv, const float* vv,
    _Float16* __restrict__ wf, float* __restrict__ cf) {
  const int idx = blockIdx.x * blockDim.x + threadIdx.x;   // 1536*256
  const int o = idx >> 8, c = idx & 255;
  const int sel = o >> 9, ow = o & 511;
  const float *w, *g, *bb, *mm, *vr;
  if (sel == 0)      { w = wq; g = gq; bb = bq; mm = mq; vr = vq; }
  else if (sel == 1) { w = wk; g = gk; bb = bk; mm = mk; vr = vk; }
  else               { w = wv; g = gv; bb = bv; mm = mv; vr = vv; }
  const float s   = g[ow] * rsqrtf(vr[ow] + 1e-5f);
  const float mul = (sel == 0) ? 0.125f : 1.0f;            // fold dk^-0.5 into q
  wf[idx] = (_Float16)(w[(size_t)ow * DIMC + c] * s * mul);
  if (c == 0) cf[o] = (bb[ow] - mm[ow] * s) * mul;
}

__global__ __launch_bounds__(256) void prep_wout(
    const float* w_out, const float* b_out, const float* go, const float* bo,
    const float* mo, const float* vo,
    _Float16* __restrict__ wof, float* __restrict__ cof) {
  const int idx = blockIdx.x * blockDim.x + threadIdx.x;   // 256*512
  const int o = idx >> 9, c = idx & 511;
  const float s = go[o] * rsqrtf(vo[o] + 1e-5f);
  wof[idx] = (_Float16)(w_out[(size_t)o * INKV + c] * s);
  if (c == 0) cof[o] = b_out[o] * s + bo[o] - mo[o] * s;
}

// ---------------- QKV GEMM: [1536 x 1024] per batch, K=256 ----------------
__global__ __launch_bounds__(256) void qkv_gemm(
    const _Float16* __restrict__ wf, const float* __restrict__ cf,
    const _Float16* __restrict__ xh,
    _Float16* __restrict__ Qo, _Float16* __restrict__ Ko, _Float16* __restrict__ VT) {
  const int lane = threadIdx.x & 31, wave = threadIdx.x >> 5;
  const int sid  = blockIdx.x * 8 + wave;                  // 24576 strips
  const int b    = sid / 1536;
  const int rem  = sid % 1536;
  const int ob   = (rem >> 4) << 4;                        // 16-wide o tile
  const int nb   = (rem & 15) << 6;                        // 64-wide n tile
  const _Float16* xb = xh + (size_t)b * NTOK * DIMC;

  v8f acc[4] = {v8f{}, v8f{}, v8f{}, v8f{}};
#pragma unroll
  for (int kb = 0; kb < DIMC; kb += 32) {
    const v16h a = load_a(wf, DIMC, ob, kb, lane);
#pragma unroll
    for (int t = 0; t < 4; ++t) {
      const v16h bt = load_b(xb, DIMC, nb + t * 16, kb, lane);
      acc[t] = wmma_f16(a, bt, acc[t]);
    }
  }

  const int hi = lane >> 4, ln = lane & 15;
  float cfr[8];
  *(float4*)&cfr[0] = *(const float4*)(cf + ob + hi * 8);
  *(float4*)&cfr[4] = *(const float4*)(cf + ob + hi * 8 + 4);

  const int typ = ob >> 9;                                 // 0:q 1:k 2:v
  const int ow  = ob & 511;
  const int h   = ow >> 6;
  const int db  = ow & 63;

  if (typ < 2) {
    _Float16* dst = (typ == 0) ? Qo : Ko;                  // [b][h][n][64]
#pragma unroll
    for (int t = 0; t < 4; ++t) {
      const int n = nb + t * 16 + ln;
      unsigned pk[4];
#pragma unroll
      for (int w = 0; w < 4; ++w)
        pk[w] = packh2(acc[t][2 * w] + cfr[2 * w], acc[t][2 * w + 1] + cfr[2 * w + 1]);
      *(uint4*)(dst + (((size_t)(b * HEADS + h) * NTOK + n) * DKV) + db + 8 * hi) =
          *(uint4*)pk;
    }
  } else {                                                 // V -> transposed [b][h][64][n]
#pragma unroll
    for (int t = 0; t < 4; ++t) {
      const int n = nb + t * 16 + ln;
#pragma unroll
      for (int r = 0; r < 8; ++r) {
        const int d = db + r + 8 * hi;
        VT[((size_t)(b * HEADS + h) * DKV + d) * NTOK + n] =
            (_Float16)(acc[t][r] + cfr[r]);
      }
    }
  }
}

// ---------------- flash attention + hardswish ----------------
// S^T = K·Q^T (lane owns query row i), Out^T = V^T·P.
// K / V^T slabs staged in LDS (async, double buffered), shared by 8 waves.
__global__ __launch_bounds__(256) void attn(
    const _Float16* __restrict__ Q, const _Float16* __restrict__ Kx,
    const _Float16* __restrict__ VT, const float* __restrict__ emb,
    const int* __restrict__ pos, _Float16* __restrict__ HS) {
  __shared__ _Float16 kbuf[2][32 * KLD];
  __shared__ _Float16 vbuf[2][64 * VLD];

  const int t    = threadIdx.x;
  const int lane = t & 31, wave = t >> 5;
  const int seg  = blockIdx.x & 7;
  const int bh   = blockIdx.x >> 3;
  const int b    = bh >> 3, h = bh & 7;
  const int ibase = seg * 128 + wave * 16;

  const _Float16* qb = Q  + (size_t)(b * HEADS + h) * NTOK * DKV;
  const _Float16* kg = Kx + (size_t)(b * HEADS + h) * NTOK * DKV;
  const _Float16* vg = VT + (size_t)(b * HEADS + h) * DKV * NTOK;

  const int hi = lane >> 4, ln = lane & 15;
  const int i  = ibase + ln;
  const int* prow = pos + (size_t)i * NTOK;

  // Loop-invariant Q B-operand (this wave's 16 query columns, both d-chunks)
  const v16h bq0 = load_b(qb, DKV, ibase, 0, lane);
  const v16h bq1 = load_b(qb, DKV, ibase, 32, lane);

  float m = -1e30f, l = 0.0f;
  v8f O[4] = {v8f{}, v8f{}, v8f{}, v8f{}};

  stage_slab(kg, vg, kbuf[0], vbuf[0], 0, t);

  int s = 0;
  for (int jb = 0; jb < NTOK; jb += 32, s ^= 1) {
#if USE_ASYNC_LDS
    __builtin_amdgcn_s_wait_asynccnt(0);
#endif
    __syncthreads();                       // slab s ready; buffer s^1 free
    if (jb + 32 < NTOK)
      stage_slab(kg, vg, kbuf[s ^ 1], vbuf[s ^ 1], jb + 32, t);

    const _Float16* ks = kbuf[s];
    const _Float16* vs = vbuf[s];

    // ---- logits: two 16x16 S^T tiles over d=64 (A from LDS) ----
    v8f s0 = v8f{}, s1 = v8f{};
    {
      v16h a0 = load_a(ks, KLD, 0,  0, lane);
      v16h a1 = load_a(ks, KLD, 16, 0, lane);
      s0 = wmma_f16(a0, bq0, s0);
      s1 = wmma_f16(a1, bq0, s1);
      a0 = load_a(ks, KLD, 0,  32, lane);
      a1 = load_a(ks, KLD, 16, 32, lane);
      s0 = wmma_f16(a0, bq1, s0);
      s1 = wmma_f16(a1, bq1, s1);
    }
    // ---- relative position bias: logits += emb[pos[i][j]][h] / SCALE ----
    float sv[16];
    {
      const int off0 = jb + 8 * hi;
      const int4 i0 = *(const int4*)(prow + off0);
      const int4 i1 = *(const int4*)(prow + off0 + 4);
      sv[0] = s0[0] + emb[i0.x * HEADS + h] * 8.0f;
      sv[1] = s0[1] + emb[i0.y * HEADS + h] * 8.0f;
      sv[2] = s0[2] + emb[i0.z * HEADS + h] * 8.0f;
      sv[3] = s0[3] + emb[i0.w * HEADS + h] * 8.0f;
      sv[4] = s0[4] + emb[i1.x * HEADS + h] * 8.0f;
      sv[5] = s0[5] + emb[i1.y * HEADS + h] * 8.0f;
      sv[6] = s0[6] + emb[i1.z * HEADS + h] * 8.0f;
      sv[7] = s0[7] + emb[i1.w * HEADS + h] * 8.0f;
      const int off1 = jb + 16 + 8 * hi;
      const int4 j0 = *(const int4*)(prow + off1);
      const int4 j1 = *(const int4*)(prow + off1 + 4);
      sv[8]  = s1[0] + emb[j0.x * HEADS + h] * 8.0f;
      sv[9]  = s1[1] + emb[j0.y * HEADS + h] * 8.0f;
      sv[10] = s1[2] + emb[j0.z * HEADS + h] * 8.0f;
      sv[11] = s1[3] + emb[j0.w * HEADS + h] * 8.0f;
      sv[12] = s1[4] + emb[j1.x * HEADS + h] * 8.0f;
      sv[13] = s1[5] + emb[j1.y * HEADS + h] * 8.0f;
      sv[14] = s1[6] + emb[j1.z * HEADS + h] * 8.0f;
      sv[15] = s1[7] + emb[j1.w * HEADS + h] * 8.0f;
    }
    // ---- streaming softmax (row i per lane; partner lane holds other j half) ----
    float mloc = sv[0];
#pragma unroll
    for (int r = 1; r < 16; ++r) mloc = fmaxf(mloc, sv[r]);
    mloc = fmaxf(mloc, __shfl_xor(mloc, 16, 32));
    const float mnew  = fmaxf(m, mloc);
    const float alpha = fexp2((m - mnew) * L2E);
    m = mnew;
    float ps = 0.0f, p0[8], p1[8];
#pragma unroll
    for (int r = 0; r < 8; ++r) { p0[r] = fexp2((sv[r]     - m) * L2E); ps += p0[r]; }
#pragma unroll
    for (int r = 0; r < 8; ++r) { p1[r] = fexp2((sv[8 + r] - m) * L2E); ps += p1[r]; }
    l = l * alpha + ps;

    // ---- pack P to f16 B-operand (needs L<->L^16 half swap) ----
    unsigned pk0[4], pk1[4], x0[4], x1[4];
#pragma unroll
    for (int w = 0; w < 4; ++w) {
      pk0[w] = packh2(p0[2 * w], p0[2 * w + 1]);
      pk1[w] = packh2(p1[2 * w], p1[2 * w + 1]);
    }
#pragma unroll
    for (int w = 0; w < 4; ++w) {
      x0[w] = __shfl_xor(pk0[w], 16, 32);
      x1[w] = __shfl_xor(pk1[w], 16, 32);
    }
    ABu bp;
#pragma unroll
    for (int w = 0; w < 4; ++w) {
      bp.u[w]     = (hi == 0) ? pk0[w] : x1[w];
      bp.u[4 + w] = (hi == 0) ? x0[w]  : pk1[w];
    }
    // ---- Out^T += V^T(slab) · P (A from LDS) ----
#pragma unroll
    for (int dt = 0; dt < 4; ++dt) {
      const v16h av = load_a(vs, VLD, dt * 16, 0, lane);
#pragma unroll
      for (int e = 0; e < 8; ++e) O[dt][e] *= alpha;
      O[dt] = wmma_f16(av, bp.v, O[dt]);
    }
  }

  // ---- normalize, hardswish, store hs[b][n][h*64+d] f16 ----
  const float lt  = l + __shfl_xor(l, 16, 32);
  const float inv = 1.0f / lt;
  _Float16* out = HS + ((size_t)b * NTOK + i) * INKV + h * DKV;
#pragma unroll
  for (int dt = 0; dt < 4; ++dt) {
    unsigned pk[4];
#pragma unroll
    for (int w = 0; w < 4; ++w) {
      float a = O[dt][2 * w] * inv;
      float c = O[dt][2 * w + 1] * inv;
      a = a * fminf(fmaxf((a + 3.0f) * (1.0f / 6.0f), 0.0f), 1.0f);
      c = c * fminf(fmaxf((c + 3.0f) * (1.0f / 6.0f), 0.0f), 1.0f);
      pk[w] = packh2(a, c);
    }
    *(uint4*)(out + dt * 16 + 8 * hi) = *(uint4*)pk;
  }
}

// ---------------- output projection: y[b][o][n] = Wof·hs + cof ----------------
__global__ __launch_bounds__(256) void proj(
    const _Float16* __restrict__ wof, const float* __restrict__ cof,
    const _Float16* __restrict__ HS, float* __restrict__ Y) {
  const int lane = threadIdx.x & 31, wave = threadIdx.x >> 5;
  const int sid  = blockIdx.x * 8 + wave;                  // 4096 strips
  const int b    = sid >> 8;
  const int rem  = sid & 255;
  const int ob   = (rem >> 4) << 4;
  const int nb   = (rem & 15) << 6;
  const _Float16* hb = HS + (size_t)b * NTOK * INKV;

  v8f acc[4] = {v8f{}, v8f{}, v8f{}, v8f{}};
#pragma unroll
  for (int kb = 0; kb < INKV; kb += 32) {
    const v16h a = load_a(wof, INKV, ob, kb, lane);
#pragma unroll
    for (int t = 0; t < 4; ++t) {
      const v16h bt = load_b(hb, INKV, nb + t * 16, kb, lane);
      acc[t] = wmma_f16(a, bt, acc[t]);
    }
  }
  const int hi = lane >> 4, ln = lane & 15;
  float cfr[8];
  *(float4*)&cfr[0] = *(const float4*)(cof + ob + hi * 8);
  *(float4*)&cfr[4] = *(const float4*)(cof + ob + hi * 8 + 4);
#pragma unroll
  for (int t = 0; t < 4; ++t) {
    const int n = nb + t * 16 + ln;
#pragma unroll
    for (int r = 0; r < 8; ++r) {
      const int o = ob + r + 8 * hi;
      Y[((size_t)b * DIMC + o) * NTOK + n] = acc[t][r] + cfr[r];
    }
  }
}

// ---------------- launch ----------------
extern "C" void kernel_launch(void* const* d_in, const int* in_sizes, int n_in,
                              void* d_out, int out_size, void* d_ws, size_t ws_size,
                              hipStream_t stream) {
  (void)in_sizes; (void)n_in; (void)out_size; (void)ws_size;
  const float* x     = (const float*)d_in[0];
  const float* wq    = (const float*)d_in[1];
  const float* gq    = (const float*)d_in[2];
  const float* bq    = (const float*)d_in[3];
  const float* mq    = (const float*)d_in[4];
  const float* vq    = (const float*)d_in[5];
  const float* wk    = (const float*)d_in[6];
  const float* gk    = (const float*)d_in[7];
  const float* bk    = (const float*)d_in[8];
  const float* mk    = (const float*)d_in[9];
  const float* vk    = (const float*)d_in[10];
  const float* wv    = (const float*)d_in[11];
  const float* gv    = (const float*)d_in[12];
  const float* bv    = (const float*)d_in[13];
  const float* mv    = (const float*)d_in[14];
  const float* vv    = (const float*)d_in[15];
  const float* emb   = (const float*)d_in[16];
  const float* w_out = (const float*)d_in[17];
  const float* b_out = (const float*)d_in[18];
  const float* go    = (const float*)d_in[19];
  const float* bo    = (const float*)d_in[20];
  const float* mo    = (const float*)d_in[21];
  const float* vo    = (const float*)d_in[22];
  const int*   pos   = (const int*)d_in[23];

  char* ws = (char*)d_ws;
  _Float16* XH  = (_Float16*)(ws + OFF_XH);
  _Float16* WF  = (_Float16*)(ws + OFF_WF);
  float*    CF  = (float*)   (ws + OFF_CF);
  _Float16* WOF = (_Float16*)(ws + OFF_WOF);
  float*    COF = (float*)   (ws + OFF_COF);
  _Float16* Qb  = (_Float16*)(ws + OFF_Q);
  _Float16* Kb  = (_Float16*)(ws + OFF_K);
  _Float16* VT  = (_Float16*)(ws + OFF_VT);
  _Float16* HS  = (_Float16*)(ws + OFF_HS);

  prep_xt<<<dim3(DIMC / 32, NTOK / 32, NB), dim3(32, 8), 0, stream>>>(x, XH);
  prep_wqkv<<<(OC3 * DIMC) / 256, 256, 0, stream>>>(
      wq, gq, bq, mq, vq, wk, gk, bk, mk, vk, wv, gv, bv, mv, vv, WF, CF);
  prep_wout<<<(DIMC * INKV) / 256, 256, 0, stream>>>(
      w_out, b_out, go, bo, mo, vo, WOF, COF);
  qkv_gemm<<<(NB * (OC3 / 16) * (NTOK / 64)) / 8, 256, 0, stream>>>(
      WF, CF, XH, Qb, Kb, VT);
  attn<<<NB * HEADS * (NTOK / 128), 256, 0, stream>>>(Qb, Kb, VT, emb, pos, HS);
  proj<<<(NB * (DIMC / 16) * (NTOK / 64)) / 8, 256, 0, stream>>>(
      WOF, COF, HS, (float*)d_out);
}
